// DAEGC_72232759984500
// MI455X (gfx1250) — matrix operand order repaired and checked by reading, not compiled, not run on hardware
//
#include <hip/hip_runtime.h>
#include <hip/hip_bf16.h>
#include <math.h>

#define N_NODES 4096
#define F_IN    1433
#define HID     256
#define EMB     16
#define KCLUS   10
#define ALPHA   0.2f
#define NEG_INF (-9e15f)

typedef __attribute__((ext_vector_type(16))) __bf16 v16bf;
typedef __attribute__((ext_vector_type(8)))  float  v8f;

// Packed fp32 -> bf16x2 conversion: single v_cvt_pk_bf16_f32 (CDNA5 VOP3).
__device__ __forceinline__ unsigned pack_bf16(float a, float b) {
    unsigned r;
    asm("v_cvt_pk_bf16_f32 %0, %1, %2" : "=v"(r) : "v"(a), "v"(b));
    return r;
}

// Fast hardware reciprocal: single v_rcp_f32 (no IEEE div expansion).
__device__ __forceinline__ float fast_rcp(float x) {
    return __builtin_amdgcn_rcpf(x);
}

// A-fragment K index for 16-bit 16x32 A tile (ISA 7.12.2):
// lane l holds M = l%16 ; element k' maps to K = 16*(k'/8) + 8*(l/16) + k'%8
__device__ __forceinline__ int a_frag_k(int kp, int half) {
    return 16 * (kp >> 3) + 8 * half + (kp & 7);
}

// -------------------------------------------------------------------------
// C[M,N] = A[M,K] @ B[K,N], fp32 in/out, bf16 WMMA fp32-accumulate.
// Compile-time K/N: constant-strength-reduced addressing, tail eliminated
// statically when KDIM % 32 == 0. LDS tiles stored in fragment order
// (lane-major packed bf16 pairs) -> each fragment is one 32-byte LDS read.
// -------------------------------------------------------------------------
template <int KDIM, int NDIM>
__global__ void __launch_bounds__(32)
gemm_bf16_wmma(const float* __restrict__ A, const float* __restrict__ B,
               float* __restrict__ C) {
    __shared__ __align__(32) unsigned lds_af[32][8];   // [lane][k'/2]
    __shared__ __align__(32) unsigned lds_bf[32][8];   // [lane][k'/2]
    const int lane = threadIdx.x;
    const int half = lane >> 4;
    const int lm   = lane & 15;
    const int m0 = blockIdx.y * 16;
    const int n0 = blockIdx.x * 16;

    // A pair (r, c..c+1) -> lane r+16*((c>>3)&1), slot ((c&7)+8*(c>>4))>>1
    // B pair (K=2q..2q+1, n) -> lane n+16*((q>>2)&1), slot (q&3)+4*(q>>3)
    constexpr int KFULL = KDIM & ~31;

    v8f acc = {};
#pragma unroll 1
    for (int k0 = 0; k0 < KFULL; k0 += 32) {
        for (int t = 0; t < 8; ++t) {
            int p = t * 32 + lane;
            int r = p >> 4;
            int c = (p & 15) * 2;
            const float* ap = A + (m0 + r) * KDIM + k0 + c;
            lds_af[r + 16 * ((c >> 3) & 1)][((c & 7) + 8 * (c >> 4)) >> 1] =
                pack_bf16(ap[0], ap[1]);
        }
        for (int t = 0; t < 8; ++t) {
            int p = t * 32 + lane;
            int n = p & 15;
            int q = p >> 4;
            const float* bp = B + (k0 + 2 * q) * NDIM + n0 + n;
            lds_bf[n + 16 * ((q >> 2) & 1)][(q & 3) + 4 * (q >> 3)] = pack_bf16(bp[0], bp[NDIM]);
        }
        __syncthreads();
        v16bf af  = *(const v16bf*)(&lds_af[lane][0]);
        v16bf bfv = *(const v16bf*)(&lds_bf[lane][0]);
        acc = __builtin_amdgcn_wmma_f32_16x16x32_bf16(false, af, false, bfv,
                                                      (short)0, acc, false, false);
        __syncthreads();
    }
    if constexpr (KFULL < KDIM) {   // tail chunk: clamp + select, branchless
        constexpr int k0 = KFULL;
        for (int t = 0; t < 8; ++t) {
            int p = t * 32 + lane;
            int r = p >> 4;
            int c = (p & 15) * 2;
            int cc0 = k0 + c, cc1 = cc0 + 1;
            const float* arow = A + (m0 + r) * KDIM;
            float t0 = arow[min(cc0, KDIM - 1)];
            float t1 = arow[min(cc1, KDIM - 1)];
            float v0 = (cc0 < KDIM) ? t0 : 0.f;
            float v1 = (cc1 < KDIM) ? t1 : 0.f;
            lds_af[r + 16 * ((c >> 3) & 1)][((c & 7) + 8 * (c >> 4)) >> 1] =
                pack_bf16(v0, v1);
        }
        for (int t = 0; t < 8; ++t) {
            int p = t * 32 + lane;
            int n = p & 15;
            int q = p >> 4;
            int r0 = k0 + 2 * q, r1 = r0 + 1;
            const float* bcol = B + n0 + n;
            float t0 = bcol[min(r0, KDIM - 1) * NDIM];
            float t1 = bcol[min(r1, KDIM - 1) * NDIM];
            float v0 = (r0 < KDIM) ? t0 : 0.f;
            float v1 = (r1 < KDIM) ? t1 : 0.f;
            lds_bf[n + 16 * ((q >> 2) & 1)][(q & 3) + 4 * (q >> 3)] = pack_bf16(v0, v1);
        }
        __syncthreads();
        v16bf af  = *(const v16bf*)(&lds_af[lane][0]);
        v16bf bfv = *(const v16bf*)(&lds_bf[lane][0]);
        acc = __builtin_amdgcn_wmma_f32_16x16x32_bf16(false, af, false, bfv,
                                                      (short)0, acc, false, false);
    }
    for (int r = 0; r < 8; ++r)
        C[(m0 + r + 8 * half) * NDIM + (n0 + lm)] = acc[r];
}

// -------------------------------------------------------------------------
// LDS-tiled transpose + fp32->bf16: hbT[d][j] = (bf16)h[j][d].
// -------------------------------------------------------------------------
template <int D>
__global__ void __launch_bounds__(256)
transpose_to_bf16(const float* __restrict__ h, __bf16* __restrict__ hbT) {
    __shared__ float tile[32][33];
    const int tx = threadIdx.x;           // 0..31
    const int ty = threadIdx.y;           // 0..7
    const int j0 = blockIdx.x * 32;       // node rows
    const int d0 = blockIdx.y * 32;       // feature cols
    for (int i = 0; i < 32; i += 8)
        tile[ty + i][tx] = (d0 + tx < D) ? h[(j0 + ty + i) * D + d0 + tx] : 0.f;
    __syncthreads();
    for (int i = 0; i < 32; i += 8)
        if (d0 + ty + i < D)
            hbT[(d0 + ty + i) * N_NODES + j0 + tx] = (__bf16)tile[tx][ty + i];
}

// -------------------------------------------------------------------------
// s_self[i] = h[i,:]·a_self ; s_neigh[i] = h[i,:]·a_neigh. One wave per row.
// -------------------------------------------------------------------------
template <int D>
__global__ void __launch_bounds__(128)
row_scores(const float* __restrict__ h, const float* __restrict__ a_self,
           const float* __restrict__ a_neigh, float* __restrict__ s_self,
           float* __restrict__ s_neigh) {
    const int wave = threadIdx.x >> 5;
    const int lane = threadIdx.x & 31;
    const int row  = blockIdx.x * 4 + wave;
    float ps = 0.f, pn = 0.f;
    for (int d = lane; d < D; d += 32) {
        float hv = h[row * D + d];
        ps += hv * a_self[d];
        pn += hv * a_neigh[d];
    }
    for (int off = 16; off > 0; off >>= 1) {
        ps += __shfl_xor(ps, off, 32);
        pn += __shfl_xor(pn, off, 32);
    }
    if (lane == 0) { s_self[row] = ps; s_neigh[row] = pn; }
}

// -------------------------------------------------------------------------
// Fused GAT attention (flash-style): logits -> leakyrelu -> mask -> online
// softmax -> acc += P @ H via WMMA -> ELU. H is pre-transposed bf16, so
// every B-fragment is one contiguous 32-byte global load.
// -------------------------------------------------------------------------
template <int D>
__global__ void __launch_bounds__(32)
gat_attention(const __bf16* __restrict__ hbT, const float* __restrict__ s_self,
              const float* __restrict__ s_neigh, const float* __restrict__ Mprox,
              const float* __restrict__ adj, float* __restrict__ out) {
    __shared__ __align__(16) float p_tile[16][32];
    __shared__ float scale_lds[16];
    __shared__ float linv_lds[16];
    const int lane = threadIdx.x;
    const int half = lane >> 4;
    const int lm   = lane & 15;
    const int i0   = blockIdx.x * 16;

    constexpr int NG = D / 16;
    v8f zero = {};
    v8f acc[NG];
    for (int g = 0; g < NG; ++g) acc[g] = zero;
    float m_run = -3.4e38f, l_run = 0.f;

#pragma unroll 1
    for (int j0 = 0; j0 < N_NODES; j0 += 32) {
        // prefetch next chunk of the N x N operands (-> global_prefetch_b8)
        if (j0 + 32 < N_NODES) {
            size_t pidx = (size_t)(i0 + lm) * N_NODES + j0 + 32;
            __builtin_prefetch(Mprox + pidx, 0, 0);
            __builtin_prefetch(adj + pidx, 0, 0);
        }
        // logits tile, paired float2 loads (coalesced along columns)
        for (int t = 0; t < 8; ++t) {
            int p = t * 32 + lane;
            int r = p >> 4;
            int c = (p & 15) * 2;
            size_t idx = (size_t)(i0 + r) * N_NODES + (j0 + c);
            float2 mv = *(const float2*)(Mprox + idx);
            float2 av = *(const float2*)(adj + idx);
            float sr = s_self[i0 + r];
            float lg0 = (sr + s_neigh[j0 + c])     * mv.x;
            float lg1 = (sr + s_neigh[j0 + c + 1]) * mv.y;
            lg0 = lg0 > 0.f ? lg0 : ALPHA * lg0;
            lg1 = lg1 > 0.f ? lg1 : ALPHA * lg1;
            float2 lg;
            lg.x = av.x > 0.f ? lg0 : NEG_INF;
            lg.y = av.y > 0.f ? lg1 : NEG_INF;
            *(float2*)&p_tile[r][c] = lg;
        }
        __syncthreads();
        // online softmax: lanes 0..15 each own one row; float4 LDS traffic
        if (lane < 16) {
            float4 vv[8];
            float cmax = -3.4e38f;
            const float4* rowp = (const float4*)&p_tile[lane][0];
            for (int c4 = 0; c4 < 8; ++c4) {
                vv[c4] = rowp[c4];
                cmax = fmaxf(cmax, fmaxf(fmaxf(vv[c4].x, vv[c4].y),
                                         fmaxf(vv[c4].z, vv[c4].w)));
            }
            float m_new = fmaxf(m_run, cmax);
            float scl   = __expf(m_run - m_new);
            float psum  = 0.f;
            float4* roww = (float4*)&p_tile[lane][0];
            for (int c4 = 0; c4 < 8; ++c4) {
                float4 e;
                e.x = __expf(vv[c4].x - m_new);
                e.y = __expf(vv[c4].y - m_new);
                e.z = __expf(vv[c4].z - m_new);
                e.w = __expf(vv[c4].w - m_new);
                psum += (e.x + e.y) + (e.z + e.w);
                roww[c4] = e;
            }
            l_run = l_run * scl + psum;
            m_run = m_new;
            scale_lds[lane] = scl;
        }
        __syncthreads();
        // rescale running accumulators
        for (int r = 0; r < 8; ++r) {
            float scl = scale_lds[r + 8 * half];
            for (int g = 0; g < NG; ++g) acc[g][r] *= scl;
        }
        // P fragment: paired reads + packed converts
        union { v16bf v; unsigned u[8]; } pf;
        for (int kp = 0; kp < 16; kp += 2) {
            float2 pv = *(const float2*)&p_tile[lm][16 * half + kp];
            pf.u[kp >> 1] = pack_bf16(pv.x, pv.y);
        }
        // acc += P @ H : one contiguous 32B load per fragment from hbT
        for (int g = 0; g < NG; ++g) {
            v16bf hf = *(const v16bf*)(hbT + (g * 16 + lm) * N_NODES
                                           + j0 + 16 * half);
            acc[g] = __builtin_amdgcn_wmma_f32_16x16x32_bf16(false, pf.v, false, hf,
                                                             (short)0, acc[g], false, false);
        }
        __syncthreads();
    }
    if (lane < 16) linv_lds[lane] = fast_rcp(l_run);   // one v_rcp per row
    __syncthreads();
    for (int g = 0; g < NG; ++g) {
        for (int r = 0; r < 8; ++r) {
            int row = r + 8 * half;
            float v = acc[g][r] * linv_lds[row];
            v = v > 0.f ? v : (__expf(v) - 1.f);          // ELU
            out[(i0 + row) * D + g * 16 + lm] = v;
        }
    }
}

// -------------------------------------------------------------------------
// z = h / max(||h||_2, 1e-12); also writes a packed bf16 copy for the decoder.
// -------------------------------------------------------------------------
__global__ void __launch_bounds__(256)
normalize_rows(const float* __restrict__ hin, float* __restrict__ z,
               __bf16* __restrict__ zb) {
    int row = blockIdx.x * blockDim.x + threadIdx.x;
    if (row >= N_NODES) return;
    float v[EMB], s = 0.f;
    for (int d = 0; d < EMB; ++d) { v[d] = hin[row * EMB + d]; s += v[d] * v[d]; }
    float inv = fast_rcp(fmaxf(sqrtf(s), 1e-12f));
    for (int d = 0; d < EMB; d += 2) {
        float z0 = v[d] * inv, z1 = v[d + 1] * inv;
        z[row * EMB + d]     = z0;
        z[row * EMB + d + 1] = z1;
        *(unsigned*)(zb + row * EMB + d) = pack_bf16(z0, z1);
    }
}

// -------------------------------------------------------------------------
// A_pred = sigmoid(z @ z^T); one wave per 16x16 tile, K=EMB=16 zero-padded
// to WMMA K=32. Fragments are contiguous 16B/32B loads of the bf16 z copy.
// -------------------------------------------------------------------------
__global__ void __launch_bounds__(32)
decoder_sigmoid(const __bf16* __restrict__ zb, float* __restrict__ A_pred) {
    const int lane = threadIdx.x;
    const int half = lane >> 4;
    const int lm   = lane & 15;
    const int i0 = blockIdx.y * 16;
    const int j0 = blockIdx.x * 16;

    union { v16bf v; unsigned u[8]; } af, bfv;
    // A-frag: k'<8 -> K = 8*half + k' in [0,16); k'>=8 -> K>=16 -> zero pad
    const unsigned* pa = (const unsigned*)(zb + (i0 + lm) * EMB + 8 * half);
    for (int i = 0; i < 4; ++i) af.u[i] = pa[i];
    for (int i = 4; i < 8; ++i) af.u[i] = 0u;
    // B-frag: k'<8 -> K = 8*half + k' in [0,16); k'>=8 -> K>=16 -> zero pad
    const unsigned* pb = (const unsigned*)(zb + (j0 + lm) * EMB + 8 * half);
    for (int i = 0; i < 4; ++i) bfv.u[i] = pb[i];
    for (int i = 4; i < 8; ++i) bfv.u[i] = 0u;
    v8f acc = {};
    acc = __builtin_amdgcn_wmma_f32_16x16x32_bf16(false, af.v, false, bfv.v,
                                                  (short)0, acc, false, false);
    for (int r = 0; r < 8; ++r) {
        float s = fast_rcp(1.f + __expf(-acc[r]));        // sigmoid, 1 v_rcp
        A_pred[(size_t)(i0 + r + 8 * half) * N_NODES + (j0 + lm)] = s;
    }
}

// -------------------------------------------------------------------------
// Student-t soft assignment; V=1 => q = (1+d2)^-1, row-normalized.
// -------------------------------------------------------------------------
__global__ void __launch_bounds__(256)
soft_assign(const float* __restrict__ z, const float* __restrict__ clus,
            float* __restrict__ q) {
    int row = blockIdx.x * blockDim.x + threadIdx.x;
    if (row >= N_NODES) return;
    float zv[EMB];
    for (int d = 0; d < EMB; ++d) zv[d] = z[row * EMB + d];
    float qs[KCLUS], tot = 0.f;
    for (int k = 0; k < KCLUS; ++k) {
        float d2 = 0.f;
        for (int d = 0; d < EMB; ++d) {
            float df = zv[d] - clus[k * EMB + d];
            d2 += df * df;
        }
        qs[k] = fast_rcp(1.f + d2);
        tot += qs[k];
    }
    float tinv = fast_rcp(tot);
    for (int k = 0; k < KCLUS; ++k) q[row * KCLUS + k] = qs[k] * tinv;
}

// -------------------------------------------------------------------------
extern "C" void kernel_launch(void* const* d_in, const int* in_sizes, int n_in,
                              void* d_out, int out_size, void* d_ws, size_t ws_size,
                              hipStream_t stream) {
    const float* x        = (const float*)d_in[0];
    const float* adj      = (const float*)d_in[1];
    const float* Mprox    = (const float*)d_in[2];
    const float* W1       = (const float*)d_in[3];
    const float* a_self1  = (const float*)d_in[4];
    const float* a_neigh1 = (const float*)d_in[5];
    const float* W2       = (const float*)d_in[6];
    const float* a_self2  = (const float*)d_in[7];
    const float* a_neigh2 = (const float*)d_in[8];
    const float* clus     = (const float*)d_in[9];

    float* A_pred = (float*)d_out;
    float* z      = A_pred + (size_t)N_NODES * N_NODES;
    float* q      = z + (size_t)N_NODES * EMB;

    float* ws    = (float*)d_ws;
    float* h1    = ws;                                  // N*HID
    float* hatt1 = h1 + (size_t)N_NODES * HID;          // N*HID
    float* h2    = hatt1 + (size_t)N_NODES * HID;       // N*EMB
    float* hatt2 = h2 + (size_t)N_NODES * EMB;          // N*EMB
    float* ssf   = hatt2 + (size_t)N_NODES * EMB;       // N
    float* snb   = ssf + N_NODES;                       // N
    __bf16* hbT  = (__bf16*)(snb + N_NODES);            // N*HID bf16 (reused)
    __bf16* zb   = hbT + (size_t)N_NODES * HID;         // N*EMB bf16

    // ---- layer 1 ----
    gemm_bf16_wmma<F_IN, HID><<<dim3(HID / 16, N_NODES / 16), 32, 0, stream>>>(
        x, W1, h1);
    row_scores<HID><<<N_NODES / 4, 128, 0, stream>>>(h1, a_self1, a_neigh1, ssf, snb);
    transpose_to_bf16<HID><<<dim3(N_NODES / 32, HID / 32), dim3(32, 8), 0, stream>>>(
        h1, hbT);
    gat_attention<HID><<<N_NODES / 16, 32, 0, stream>>>(hbT, ssf, snb, Mprox, adj, hatt1);

    // ---- layer 2 ----
    gemm_bf16_wmma<HID, EMB><<<dim3(EMB / 16, N_NODES / 16), 32, 0, stream>>>(
        hatt1, W2, h2);
    row_scores<EMB><<<N_NODES / 4, 128, 0, stream>>>(h2, a_self2, a_neigh2, ssf, snb);
    transpose_to_bf16<EMB><<<dim3(N_NODES / 32, 1), dim3(32, 8), 0, stream>>>(
        h2, hbT);
    gat_attention<EMB><<<N_NODES / 16, 32, 0, stream>>>(hbT, ssf, snb, Mprox, adj, hatt2);

    // ---- outputs ----
    normalize_rows<<<N_NODES / 256, 256, 0, stream>>>(hatt2, z, zb);
    decoder_sigmoid<<<dim3(N_NODES / 16, N_NODES / 16), 32, 0, stream>>>(zb, A_pred);
    soft_assign<<<N_NODES / 256, 256, 0, stream>>>(z, clus, q);
}
